// Attention_26723286516133
// MI455X (gfx1250) — matrix operand order repaired and checked by reading, not compiled, not run on hardware
//
#include <hip/hip_runtime.h>

// ---------------------------------------------------------------------------
// Types for CDNA5 WMMA (wave32, v_wmma_f32_16x16x32_bf16)
// ---------------------------------------------------------------------------
typedef __bf16 bf16;
typedef bf16  v16bf __attribute__((ext_vector_type(16)));
typedef bf16  v8bf  __attribute__((ext_vector_type(8)));
typedef float v8f   __attribute__((ext_vector_type(8)));

static __device__ __forceinline__ v16bf cat8(v8bf lo, v8bf hi) {
  v16bf r;
#pragma unroll
  for (int i = 0; i < 8; ++i) { r[i] = lo[i]; r[i + 8] = hi[i]; }
  return r;
}

static __device__ __forceinline__ v8f wmma_bf16(v16bf a, v16bf b, v8f c) {
  // D = A(16x32 bf16) * B(32x16 bf16) + C(16x16 f32)
  return __builtin_amdgcn_wmma_f32_16x16x32_bf16(
      false, a, false, b, (short)0, c, false, false);
}

// Problem constants
#define SEQ     2048
#define DEMB    1024
#define NHEADS  16
#define DHEAD   64

// ---------------------------------------------------------------------------
// f32 -> bf16 converts
// ---------------------------------------------------------------------------
__global__ void cvt_kernel(const float* __restrict__ in, bf16* __restrict__ out, int n) {
  int i = blockIdx.x * blockDim.x + threadIdx.x;
  if (i < n) out[i] = (bf16)in[i];
}

// W_O[a][e][h] -> W2t[e][a*64+h]   (so B-operand columns load contiguously)
__global__ void cvt_wo_kernel(const float* __restrict__ wo, bf16* __restrict__ w2t) {
  int i = blockIdx.x * blockDim.x + threadIdx.x;  // n = 16*1024*64 = 1<<20
  int a = i >> 16;
  int e = (i >> 6) & 1023;
  int h = i & 63;
  w2t[(size_t)e * 1024 + a * 64 + h] = (bf16)wo[i];
}

// ---------------------------------------------------------------------------
// QKV projection: per head a:  P[c][h] = sum_e x[c][e] * W[a][h][e]
// One wave computes a 64(M=c) x 64(N=h) tile (16 f32 accumulators);
// K-loop over e (1024, step 32). Each B fragment feeds 4 WMMAs.
//   A  = x tile (row-major over e -> two contiguous 16B chunks per lane)
//   B  = W[a]^T, i.e. B[e][h]: lane column h = l&15, 16 contiguous e.
// z=0: Q (scaled by 1/8, row-major [a][c][h])
// z=1: K (row-major [a][c][h])
// z=2: V (stored TRANSPOSED: Vt[a][h][c] so P.V columns load contiguously)
// ---------------------------------------------------------------------------
__global__ __launch_bounds__(32) void qkv_kernel(
    const bf16* __restrict__ xb,
    const bf16* __restrict__ wq, const bf16* __restrict__ wk,
    const bf16* __restrict__ wv,
    bf16* __restrict__ Qb, bf16* __restrict__ Kb, bf16* __restrict__ Vt)
{
  const int l  = threadIdx.x;
  const int ll = l & 15, lh = l >> 4;
  const int c0 = blockIdx.x * 64;
  const int a  = blockIdx.y;
  const int zs = blockIdx.z;

  const bf16* w = (zs == 0) ? wq : (zs == 1) ? wk : wv;
  w += (size_t)a * DHEAD * DEMB;

  v8f acc[16] = {};  // [u(4 row subtiles)][t(4 col subtiles)]
  const bf16* arow[4];
#pragma unroll
  for (int u = 0; u < 4; ++u) arow[u] = xb + (size_t)(c0 + 16 * u + ll) * DEMB;

  for (int e0 = 0; e0 < DEMB; e0 += 32) {
    __builtin_prefetch(arow[0] + e0 + 128, 0, 1);  // global_prefetch_b8
    v16bf af[4];
#pragma unroll
    for (int u = 0; u < 4; ++u)
      af[u] = cat8(*(const v8bf*)(arow[u] + e0 + 16 * lh),
                   *(const v8bf*)(arow[u] + e0 + 8 + 16 * lh));
#pragma unroll
    for (int t = 0; t < 4; ++t) {
      const bf16* bp = w + (size_t)(16 * t + ll) * DEMB + e0 + 16 * lh;
      v16bf bf = *(const v16bf*)bp;
#pragma unroll
      for (int u = 0; u < 4; ++u)
        acc[u * 4 + t] = wmma_bf16(af[u], bf, acc[u * 4 + t]);
    }
  }

  if (zs == 0) {
    bf16* out = Qb + (size_t)a * SEQ * DHEAD;
#pragma unroll
    for (int u = 0; u < 4; ++u)
#pragma unroll
      for (int t = 0; t < 4; ++t)
#pragma unroll
        for (int r = 0; r < 8; ++r)
          out[(size_t)(c0 + 16 * u + r + 8 * lh) * DHEAD + 16 * t + ll] =
              (bf16)(acc[u * 4 + t][r] * 0.125f);  // fold 1/sqrt(64)
  } else if (zs == 1) {
    bf16* out = Kb + (size_t)a * SEQ * DHEAD;
#pragma unroll
    for (int u = 0; u < 4; ++u)
#pragma unroll
      for (int t = 0; t < 4; ++t)
#pragma unroll
        for (int r = 0; r < 8; ++r)
          out[(size_t)(c0 + 16 * u + r + 8 * lh) * DHEAD + 16 * t + ll] =
              (bf16)acc[u * 4 + t][r];
  } else {
    bf16* out = Vt + (size_t)a * DHEAD * SEQ;  // [h][c]
#pragma unroll
    for (int u = 0; u < 4; ++u)
#pragma unroll
      for (int t = 0; t < 4; ++t)
#pragma unroll
        for (int r = 0; r < 8; ++r)
          out[(size_t)(16 * t + ll) * SEQ + c0 + 16 * u + r + 8 * lh] =
              (bf16)acc[u * 4 + t][r];
  }
}

// ---------------------------------------------------------------------------
// Flash attention (reference has Q/K roles swapped: S[c][C] = k_c . q_C, C<=c)
//   Arows = K-projection [a][c][64]  (row/output positions)
//   Bcols = Q-projection [a][C][64]  (attended positions, pre-scaled)
//   Vt    = [a][64][2048]
//   Zb    = [c][a*64+h] bf16
// One wave per 32-row tile per head (2 row subtiles); online softmax over
// 32-key steps: 16 WMMAs per step, B/V fragments reused across row subtiles.
// ---------------------------------------------------------------------------
__global__ __launch_bounds__(32) void flash_kernel(
    const bf16* __restrict__ Arows, const bf16* __restrict__ Bcols,
    const bf16* __restrict__ Vt, bf16* __restrict__ Zb)
{
  __shared__ bf16 lp[32 * 32];  // P tile staging (C-layout -> A-layout)

  const int l  = threadIdx.x;
  const int ll = l & 15, lh = l >> 4;
  const int a  = blockIdx.y;
  const int q0 = blockIdx.x * 32;

  // A fragments for the 32x64 row tile (K-dim = head dim; [u][2 frags])
  v16bf aR[2][2];
#pragma unroll
  for (int u = 0; u < 2; ++u) {
    const bf16* ar = Arows + ((size_t)a * SEQ + q0 + 16 * u + ll) * DHEAD;
    aR[u][0] = cat8(*(const v8bf*)(ar + 16 * lh),      *(const v8bf*)(ar + 8 + 16 * lh));
    aR[u][1] = cat8(*(const v8bf*)(ar + 32 + 16 * lh), *(const v8bf*)(ar + 40 + 16 * lh));
  }

  float m[2][8], lsum[2][8];
#pragma unroll
  for (int u = 0; u < 2; ++u)
#pragma unroll
    for (int r = 0; r < 8; ++r) { m[u][r] = -3.0e38f; lsum[u][r] = 0.0f; }
  v8f o[2][4] = {};

  const bf16* vb = Vt + (size_t)a * DHEAD * SEQ;
  const int kend = q0 + 32;

  for (int k0 = 0; k0 < kend; k0 += 32) {
    // ---- S = rowTile . colTile^T : B fragments shared by both row subtiles
    const bf16* b0 = Bcols + ((size_t)a * SEQ + k0 + ll) * DHEAD;
    const bf16* b1 = b0 + 16 * DHEAD;
    v16bf bq[2][2];
    bq[0][0] = *(const v16bf*)(b0 + 16 * lh);
    bq[0][1] = *(const v16bf*)(b0 + 32 + 16 * lh);
    bq[1][0] = *(const v16bf*)(b1 + 16 * lh);
    bq[1][1] = *(const v16bf*)(b1 + 32 + 16 * lh);

    v8f s[2][2] = {};  // [u row subtile][n key subtile]
#pragma unroll
    for (int u = 0; u < 2; ++u)
#pragma unroll
      for (int n = 0; n < 2; ++n) {
        s[u][n] = wmma_bf16(aR[u][0], bq[n][0], s[u][n]);
        s[u][n] = wmma_bf16(aR[u][1], bq[n][1], s[u][n]);
      }

    // ---- causal mask + online softmax (row = 16u + r + 8*lh, col = ll)
#pragma unroll
    for (int u = 0; u < 2; ++u)
#pragma unroll
      for (int r = 0; r < 8; ++r) {
        const int q = q0 + 16 * u + r + 8 * lh;
        float v0 = (k0 + ll      <= q) ? s[u][0][r] : -3.0e38f;
        float v1 = (k0 + 16 + ll <= q) ? s[u][1][r] : -3.0e38f;
        float mx = fmaxf(v0, v1);
        mx = fmaxf(mx, __shfl_xor(mx, 1, 32));
        mx = fmaxf(mx, __shfl_xor(mx, 2, 32));
        mx = fmaxf(mx, __shfl_xor(mx, 4, 32));
        mx = fmaxf(mx, __shfl_xor(mx, 8, 32));
        const float mn = fmaxf(m[u][r], mx);
        const float al = __expf(m[u][r] - mn);
        m[u][r] = mn;
        const float e0v = __expf(v0 - mn);
        const float e1v = __expf(v1 - mn);
        float rs = e0v + e1v;
        rs += __shfl_xor(rs, 1, 32);
        rs += __shfl_xor(rs, 2, 32);
        rs += __shfl_xor(rs, 4, 32);
        rs += __shfl_xor(rs, 8, 32);
        lsum[u][r] = lsum[u][r] * al + rs;
        o[u][0][r] *= al; o[u][1][r] *= al; o[u][2][r] *= al; o[u][3][r] *= al;
        // stash P (bf16) to LDS in row-major [32][32]
        lp[(16 * u + r + 8 * lh) * 32 + ll]      = (bf16)e0v;
        lp[(16 * u + r + 8 * lh) * 32 + 16 + ll] = (bf16)e1v;
      }
    asm volatile("s_wait_dscnt 0x0" ::: "memory");

    // reload P in A-layout (two 16x32 tiles, K-dim = key chunk)
    v16bf aP[2];
#pragma unroll
    for (int u = 0; u < 2; ++u)
      aP[u] = cat8(*(const v8bf*)(lp + (16 * u + ll) * 32 + 16 * lh),
                   *(const v8bf*)(lp + (16 * u + ll) * 32 + 8 + 16 * lh));

    // ---- O += P . V   (V fragments shared by both row subtiles)
#pragma unroll
    for (int t = 0; t < 4; ++t) {
      const bf16* vp = vb + (size_t)(16 * t + ll) * SEQ + k0 + 16 * lh;
      v16bf vf = *(const v16bf*)vp;
      o[0][t] = wmma_bf16(aP[0], vf, o[0][t]);
      o[1][t] = wmma_bf16(aP[1], vf, o[1][t]);
    }
  }

  // ---- normalize and emit Z[c][a*64+h]
#pragma unroll
  for (int u = 0; u < 2; ++u)
#pragma unroll
    for (int r = 0; r < 8; ++r) lsum[u][r] = 1.0f / lsum[u][r];
#pragma unroll
  for (int u = 0; u < 2; ++u)
#pragma unroll
    for (int t = 0; t < 4; ++t)
#pragma unroll
      for (int r = 0; r < 8; ++r)
        Zb[(size_t)(q0 + 16 * u + r + 8 * lh) * DEMB + a * DHEAD + 16 * t + ll] =
            (bf16)(o[u][t][r] * lsum[u][r]);
}

// ---------------------------------------------------------------------------
// Output projection: out[c][e] = sum_k Zb[c][k] * W2t[e][k], f32 output.
// One wave per 64x64 tile; K-loop over 1024 step 32; B reused by 4 WMMAs.
// ---------------------------------------------------------------------------
__global__ __launch_bounds__(32) void proj_kernel(
    const bf16* __restrict__ Zb, const bf16* __restrict__ W2t,
    float* __restrict__ out)
{
  const int l  = threadIdx.x;
  const int ll = l & 15, lh = l >> 4;
  const int c0 = blockIdx.x * 64;
  const int e0 = blockIdx.y * 64;

  v8f acc[16] = {};
  const bf16* arow[4];
#pragma unroll
  for (int u = 0; u < 4; ++u) arow[u] = Zb + (size_t)(c0 + 16 * u + ll) * DEMB;

  for (int k0 = 0; k0 < DEMB; k0 += 32) {
    __builtin_prefetch(arow[0] + k0 + 128, 0, 1);
    v16bf af[4];
#pragma unroll
    for (int u = 0; u < 4; ++u)
      af[u] = cat8(*(const v8bf*)(arow[u] + k0 + 16 * lh),
                   *(const v8bf*)(arow[u] + k0 + 8 + 16 * lh));
#pragma unroll
    for (int t = 0; t < 4; ++t) {
      const bf16* bp = W2t + (size_t)(e0 + 16 * t + ll) * DEMB + k0 + 16 * lh;
      v16bf bf = *(const v16bf*)bp;
#pragma unroll
      for (int u = 0; u < 4; ++u)
        acc[u * 4 + t] = wmma_bf16(af[u], bf, acc[u * 4 + t]);
    }
  }

#pragma unroll
  for (int u = 0; u < 4; ++u)
#pragma unroll
    for (int t = 0; t < 4; ++t)
#pragma unroll
      for (int r = 0; r < 8; ++r)
        out[(size_t)(c0 + 16 * u + r + 8 * lh) * DEMB + e0 + 16 * t + ll] =
            acc[u * 4 + t][r];
}

// ---------------------------------------------------------------------------
// Host launcher
// ---------------------------------------------------------------------------
extern "C" void kernel_launch(void* const* d_in, const int* in_sizes, int n_in,
                              void* d_out, int out_size, void* d_ws, size_t ws_size,
                              hipStream_t stream) {
  (void)in_sizes; (void)n_in; (void)out_size; (void)ws_size;
  const float* x  = (const float*)d_in[0];
  const float* Wq = (const float*)d_in[1];
  const float* Wk = (const float*)d_in[2];
  const float* Wv = (const float*)d_in[3];
  const float* Wo = (const float*)d_in[4];
  float* out = (float*)d_out;

  // bf16 workspace carve-up (~29.4 MB total)
  bf16* W = (bf16*)d_ws;
  size_t off = 0;
  bf16* xb  = W + off; off += (size_t)SEQ * DEMB;            // 2M
  bf16* wq  = W + off; off += (size_t)NHEADS * DHEAD * DEMB; // 1M
  bf16* wk  = W + off; off += (size_t)NHEADS * DHEAD * DEMB;
  bf16* wv  = W + off; off += (size_t)NHEADS * DHEAD * DEMB;
  bf16* w2t = W + off; off += (size_t)DEMB * DEMB;           // 1M
  bf16* Qb  = W + off; off += (size_t)NHEADS * SEQ * DHEAD;  // 2M
  bf16* Kb  = W + off; off += (size_t)NHEADS * SEQ * DHEAD;
  bf16* Vt  = W + off; off += (size_t)NHEADS * SEQ * DHEAD;
  bf16* Zb  = W + off; off += (size_t)SEQ * DEMB;

  const int nx = SEQ * DEMB;                 // 2,097,152
  const int nw = NHEADS * DHEAD * DEMB;      // 1,048,576
  cvt_kernel<<<dim3(nx / 256), dim3(256), 0, stream>>>(x,  xb, nx);
  cvt_kernel<<<dim3(nw / 256), dim3(256), 0, stream>>>(Wq, wq, nw);
  cvt_kernel<<<dim3(nw / 256), dim3(256), 0, stream>>>(Wk, wk, nw);
  cvt_kernel<<<dim3(nw / 256), dim3(256), 0, stream>>>(Wv, wv, nw);
  cvt_wo_kernel<<<dim3(nw / 256), dim3(256), 0, stream>>>(Wo, w2t);

  // QKV projections: 64-row tiles x 16 heads x {Q,K,V}
  qkv_kernel<<<dim3(SEQ / 64, NHEADS, 3), dim3(32), 0, stream>>>(
      xb, wq, wk, wv, Qb, Kb, Vt);

  // Attention: rows from K-projection, columns from (scaled) Q-projection
  flash_kernel<<<dim3(SEQ / 32, NHEADS), dim3(32), 0, stream>>>(Kb, Qb, Vt, Zb);

  // Output projection -> f32 d_out
  proj_kernel<<<dim3(SEQ / 64, DEMB / 64), dim3(32), 0, stream>>>(Zb, w2t, out);
}
